// WindowAttention_27135603376348
// MI455X (gfx1250) — matrix operand order repaired and checked by reading, not compiled, not run on hardware
//
#include <hip/hip_runtime.h>

#define NWIN   49
#define NHEADS 12
#define HDIM   32
#define CDIM   384
#define NQKV   1152
#define BWIN   4096
#define MROWS  (BWIN * NWIN)   // 200704, divisible by 16
#define MTILES (MROWS / 16)    // 12544
#define KTPAD  64              // padded key-row length in kT

typedef __attribute__((ext_vector_type(16))) _Float16 v16h;
typedef __attribute__((ext_vector_type(8)))  _Float16 v8h;
typedef __attribute__((ext_vector_type(8)))  float    v8f;

static __device__ __forceinline__ v16h cat8(v8h lo, v8h hi) {
  return __builtin_shufflevector(lo, hi, 0,1,2,3,4,5,6,7,8,9,10,11,12,13,14,15);
}
static __device__ __forceinline__ v8f zero8f() {
  v8f z;
#pragma unroll
  for (int i = 0; i < 8; ++i) z[i] = 0.0f;
  return z;
}
static __device__ __forceinline__ v16h zero16h() {
  v16h z;
#pragma unroll
  for (int i = 0; i < 16; ++i) z[i] = (_Float16)0.0f;
  return z;
}

// ---------------------------------------------------------------------------
// Kernel 0: convert weights to f16 (transposed so WMMA B-fragments are
// contiguous 32B loads), build padded bias table biasC[h][col][row] (64x64,
// zero pad), and zero the m=49..63 pad columns of kT.
// ---------------------------------------------------------------------------
__global__ void init_kernel(const float* __restrict__ qkv_w,
                            const float* __restrict__ proj_w,
                            const float* __restrict__ rpb,
                            _Float16* __restrict__ qkv_wt,   // [384][1152]
                            _Float16* __restrict__ proj_wt,  // [384][384]
                            float* __restrict__ biasC,       // [12][64][64]
                            _Float16* __restrict__ kT) {     // [BH*32][64]
  int idx    = blockIdx.x * blockDim.x + threadIdx.x;
  int stride = gridDim.x * blockDim.x;
  for (int t = idx; t < CDIM * NQKV; t += stride) {
    int k = t / NQKV, j = t - k * NQKV;
    qkv_wt[t] = (_Float16)qkv_w[j * CDIM + k];
  }
  for (int t = idx; t < CDIM * CDIM; t += stride) {
    int k = t / CDIM, j = t - k * CDIM;
    proj_wt[t] = (_Float16)proj_w[j * CDIM + k];
  }
  // biasC[h][m(col)][n(row)], zero outside 49x49
  for (int t = idx; t < NHEADS * 64 * 64; t += stride) {
    int h  = t >> 12;
    int rm = t & 4095;
    int m  = rm >> 6, n = rm & 63;
    float v = 0.0f;
    if (n < NWIN && m < NWIN) {
      int ni = n / 7, nj = n - ni * 7;
      int mi = m / 7, mj = m - mi * 7;
      v = rpb[((ni - mi + 6) * 13 + (nj - mj + 6)) * NHEADS + h];
    }
    biasC[t] = v;
  }
  // zero kT pad region (m = 49..63 of every (head,d) row)
  const int rows = BWIN * NHEADS * HDIM;
  for (int t = idx; t < rows * (KTPAD - NWIN); t += stride) {
    int r = t / (KTPAD - NWIN);
    int m = NWIN + (t - r * (KTPAD - NWIN));
    kT[(size_t)r * KTPAD + m] = (_Float16)0.0f;
  }
}

// ---------------------------------------------------------------------------
// Kernel 1: QKV GEMM.  out[M,1152] = x[M,384] @ qkv_w^T + qkv_b
// One block = one 16-row M tile (x staged to LDS as f16, read once from HBM).
// 8 waves x 9 N-tiles = 72 N-tiles (1152 cols).  K = 12 chunks of 32.
// Epilogue scatters q (pre-scaled), v into [head][n][d] and k into padded
// transposed kT[head][d][64].
// ---------------------------------------------------------------------------
__global__ __launch_bounds__(256) void qkv_gemm(
    const float* __restrict__ x, const _Float16* __restrict__ wt,
    const float* __restrict__ qb, _Float16* __restrict__ qd,
    _Float16* __restrict__ kT, _Float16* __restrict__ vd) {
  __shared__ __attribute__((aligned(16))) _Float16 As[16][392];
  const int mtile = blockIdx.x;
  const int tid   = threadIdx.x;
  for (int t = tid; t < 16 * CDIM; t += 256) {
    int r = t / CDIM, c = t - r * CDIM;
    As[r][c] = (_Float16)x[(size_t)(mtile * 16 + r) * CDIM + c];
  }
  __syncthreads();

  const int w = tid >> 5, lane = tid & 31;
  const int lrow = lane & 15, halfid = lane >> 4, base = halfid * 8;

  v8f acc[9];
#pragma unroll
  for (int i = 0; i < 9; ++i) acc[i] = zero8f();

  for (int kc = 0; kc < 12; ++kc) {
    const int k0 = kc * 32;
    v8h alo = *(const v8h*)&As[lrow][k0 + base];
    v8h ahi = *(const v8h*)&As[lrow][k0 + 16 + base];
    v16h a  = cat8(alo, ahi);
    const _Float16* brow = wt + (size_t)k0 * NQKV + lrow;
#pragma unroll
    for (int i = 0; i < 9; ++i) {
      v16h b;
#pragma unroll
      for (int s = 0; s < 16; ++s)
        b[s] = brow[(size_t)(base + (s & 7) + ((s >> 3) << 4)) * NQKV + (w * 9 + i) * 16];
      acc[i] = __builtin_amdgcn_wmma_f32_16x16x32_f16(
          false, a, false, b, (short)0, acc[i], false, false);
    }
  }

  const float scale = 0.17677669529663687f;  // 32^-0.5
#pragma unroll
  for (int i = 0; i < 9; ++i) {
    int j = (w * 9 + i) * 16 + lrow;
    float bj = qb[j];
    int s = j / CDIM;
    int r = j - s * CDIM;
    int hh = r >> 5, d = r & 31;
#pragma unroll
    for (int v = 0; v < 8; ++v) {
      int gm = mtile * 16 + halfid * 8 + v;
      int bw = gm / NWIN;
      int n  = gm - bw * NWIN;
      float val = acc[i][v] + bj;
      if (s == 0) {
        qd[((size_t)(bw * NHEADS + hh) * NWIN + n) * HDIM + d] =
            (_Float16)(val * scale);
      } else if (s == 1) {
        kT[((size_t)(bw * NHEADS + hh) * HDIM + d) * KTPAD + n] =
            (_Float16)val;
      } else {
        vd[((size_t)(bw * NHEADS + hh) * NWIN + n) * HDIM + d] =
            (_Float16)val;
      }
    }
  }
}

// ---------------------------------------------------------------------------
// Kernel 2: attention per (b, h).  1 block = 4 waves = 1 head.
// S = q@k^T from padded kT (single K=32 WMMA chunk per tile, vector loads),
// vectorized bias add + mask + softmax, P via LDS, ctx = P@v.
// ---------------------------------------------------------------------------
__global__ __launch_bounds__(128) void attn_kernel(
    const _Float16* __restrict__ qd, const _Float16* __restrict__ kT,
    const _Float16* __restrict__ vd, const float* __restrict__ biasC,
    _Float16* __restrict__ ctx) {
  __shared__ __attribute__((aligned(16))) _Float16 Ps[64][72];
  const int hb = blockIdx.x;
  const int bw = hb / NHEADS;
  const int h  = hb - bw * NHEADS;
  const _Float16* qh  = qd + (size_t)hb * NWIN * HDIM;
  const _Float16* kth = kT + (size_t)hb * HDIM * KTPAD;
  const _Float16* vh  = vd + (size_t)hb * NWIN * HDIM;
  const float*    bc  = biasC + h * 4096;

  const int w = threadIdx.x >> 5, lane = threadIdx.x & 31;
  const int lrow = lane & 15, halfid = lane >> 4, base = halfid * 8;
  const int rt = w;  // row tile 0..3

  // A fragment from q rows (K = hd = 32, exactly one chunk)
  v16h aq = zero16h();
  {
    int arow = rt * 16 + lrow;
    if (arow < NWIN) {
      v8h lo = *(const v8h*)(qh + arow * HDIM + base);
      v8h hi = *(const v8h*)(qh + arow * HDIM + 16 + base);
      aq = cat8(lo, hi);
    }
  }

  v8f S[4];
#pragma unroll
  for (int ct = 0; ct < 4; ++ct) {
    // B lane = column m = ct*16+lrow; elements = d in table order
    v16h bk;
#pragma unroll
    for (int s = 0; s < 16; ++s)
      bk[s] = kth[(base + (s & 7) + ((s >> 3) << 4)) * KTPAD + ct * 16 + lrow];
    S[ct] = __builtin_amdgcn_wmma_f32_16x16x32_f16(
        false, aq, false, bk, (short)0, zero8f(), false, false);
  }

  // bias (vector load) + mask
#pragma unroll
  for (int ct = 0; ct < 4; ++ct) {
    int col = ct * 16 + lrow;
    v8f bias = *(const v8f*)(bc + col * 64 + rt * 16 + halfid * 8);
    bool colv = col < NWIN;
#pragma unroll
    for (int v = 0; v < 8; ++v) {
      int row = rt * 16 + halfid * 8 + v;
      float s = S[ct][v] + bias[v];
      if (!colv || row >= NWIN) s = -1e30f;
      S[ct][v] = s;
    }
  }

  // row softmax: rows live at fixed (v, lane-half); reduce across 16 lanes
#pragma unroll
  for (int v = 0; v < 8; ++v) {
    float mx = fmaxf(fmaxf(S[0][v], S[1][v]), fmaxf(S[2][v], S[3][v]));
    mx = fmaxf(mx, __shfl_xor(mx, 1));
    mx = fmaxf(mx, __shfl_xor(mx, 2));
    mx = fmaxf(mx, __shfl_xor(mx, 4));
    mx = fmaxf(mx, __shfl_xor(mx, 8));
    float sum = 0.0f;
#pragma unroll
    for (int ct = 0; ct < 4; ++ct) {
      float p = __expf(S[ct][v] - mx);
      S[ct][v] = p;
      sum += p;
    }
    sum += __shfl_xor(sum, 1);
    sum += __shfl_xor(sum, 2);
    sum += __shfl_xor(sum, 4);
    sum += __shfl_xor(sum, 8);
    float inv = 1.0f / sum;
#pragma unroll
    for (int ct = 0; ct < 4; ++ct) S[ct][v] *= inv;
  }

  // P -> LDS (f16), rows disjoint per wave
#pragma unroll
  for (int ct = 0; ct < 4; ++ct)
#pragma unroll
    for (int v = 0; v < 8; ++v)
      Ps[rt * 16 + halfid * 8 + v][ct * 16 + lrow] = (_Float16)S[ct][v];
  __syncthreads();

  // ctx = P @ v : K = 64 (2 chunks of 32), 2 d-tiles
  v8f O[2];
  O[0] = zero8f(); O[1] = zero8f();
#pragma unroll
  for (int kc = 0; kc < 2; ++kc) {
    int k0 = kc * 32;
    int prow = rt * 16 + lrow;
    v8h lo = *(const v8h*)&Ps[prow][k0 + base];
    v8h hi = *(const v8h*)&Ps[prow][k0 + 16 + base];
    v16h ap = cat8(lo, hi);
#pragma unroll
    for (int dt = 0; dt < 2; ++dt) {
      v16h bv = zero16h();
#pragma unroll
      for (int s = 0; s < 16; ++s) {
        int m = k0 + base + (s & 7) + ((s >> 3) << 4);
        if (m < NWIN) bv[s] = vh[m * HDIM + dt * 16 + lrow];
      }
      O[dt] = __builtin_amdgcn_wmma_f32_16x16x32_f16(
          false, ap, false, bv, (short)0, O[dt], false, false);
    }
  }

  // store ctx[(b*49+row)*384 + h*32 + d] as f16 (input to proj GEMM)
#pragma unroll
  for (int dt = 0; dt < 2; ++dt) {
#pragma unroll
    for (int v = 0; v < 8; ++v) {
      int row = rt * 16 + halfid * 8 + v;
      if (row < NWIN) {
        int d = dt * 16 + lrow;
        ctx[((size_t)(bw * NWIN + row)) * CDIM + h * HDIM + d] =
            (_Float16)O[dt][v];
      }
    }
  }
}

// ---------------------------------------------------------------------------
// Kernel 3: proj GEMM. out[M,384] = ctx[M,384] @ proj_w^T + proj_b (fp32 out)
// ctx is already f16, so stage the A tile into LDS with the gfx1250 async
// LDS-load path (ASYNCcnt), bypassing the VGPR round-trip.
// ---------------------------------------------------------------------------
__global__ __launch_bounds__(256) void proj_gemm(
    const _Float16* __restrict__ ctx, const _Float16* __restrict__ wt,
    const float* __restrict__ pb, float* __restrict__ out) {
  __shared__ __attribute__((aligned(16))) _Float16 As[16][392];
  const int mtile = blockIdx.x;
  const int tid   = threadIdx.x;
  {
    // 16 rows x 384 halfs = 768 chunks of 16B; 3 chunks per thread
    const _Float16* gbase = ctx + (size_t)mtile * 16 * CDIM;
#pragma unroll
    for (int i = 0; i < 3; ++i) {
      int c  = tid + 256 * i;
      int r  = c / 48, cc = c - r * 48;
      const _Float16* gp = gbase + r * CDIM + cc * 8;
      unsigned lofs = (unsigned)(uintptr_t)(&As[r][cc * 8]);
      asm volatile("global_load_async_to_lds_b128 %0, %1, off"
                   :
                   : "v"(lofs), "v"(gp)
                   : "memory");
    }
    asm volatile("s_wait_asynccnt 0" ::: "memory");
  }
  __syncthreads();

  const int w = tid >> 5, lane = tid & 31;
  const int lrow = lane & 15, halfid = lane >> 4, base = halfid * 8;

  v8f acc[3];
#pragma unroll
  for (int i = 0; i < 3; ++i) acc[i] = zero8f();

  for (int kc = 0; kc < 12; ++kc) {
    const int k0 = kc * 32;
    v8h alo = *(const v8h*)&As[lrow][k0 + base];
    v8h ahi = *(const v8h*)&As[lrow][k0 + 16 + base];
    v16h a  = cat8(alo, ahi);
    const _Float16* brow = wt + (size_t)k0 * CDIM + lrow;
#pragma unroll
    for (int i = 0; i < 3; ++i) {
      v16h b;
#pragma unroll
      for (int s = 0; s < 16; ++s)
        b[s] = brow[(size_t)(base + (s & 7) + ((s >> 3) << 4)) * CDIM + (w * 3 + i) * 16];
      acc[i] = __builtin_amdgcn_wmma_f32_16x16x32_f16(
          false, a, false, b, (short)0, acc[i], false, false);
    }
  }

#pragma unroll
  for (int i = 0; i < 3; ++i) {
    int j = (w * 3 + i) * 16 + lrow;
    float bj = pb[j];
#pragma unroll
    for (int v = 0; v < 8; ++v) {
      int gm = mtile * 16 + halfid * 8 + v;
      out[(size_t)gm * CDIM + j] = acc[i][v] + bj;
    }
  }
}

// ---------------------------------------------------------------------------
extern "C" void kernel_launch(void* const* d_in, const int* in_sizes, int n_in,
                              void* d_out, int out_size, void* d_ws,
                              size_t ws_size, hipStream_t stream) {
  (void)in_sizes; (void)n_in; (void)out_size; (void)ws_size;
  const float* x      = (const float*)d_in[0];
  const float* qkv_w  = (const float*)d_in[1];
  const float* qkv_b  = (const float*)d_in[2];
  const float* rpb    = (const float*)d_in[3];
  const float* proj_w = (const float*)d_in[4];
  const float* proj_b = (const float*)d_in[5];
  float* out = (float*)d_out;

  char* ws = (char*)d_ws;
  size_t off = 0;
  auto carve = [&](size_t bytes) {
    void* p = ws + off;
    off = (off + bytes + 255) & ~(size_t)255;
    return p;
  };
  _Float16* qkv_wt  = (_Float16*)carve((size_t)CDIM * NQKV * 2);
  _Float16* proj_wt = (_Float16*)carve((size_t)CDIM * CDIM * 2);
  float*    biasC   = (float*)carve((size_t)NHEADS * 64 * 64 * 4);
  _Float16* qd  = (_Float16*)carve((size_t)BWIN * NHEADS * NWIN * HDIM * 2);
  _Float16* kT  = (_Float16*)carve((size_t)BWIN * NHEADS * HDIM * KTPAD * 2);
  _Float16* vd  = (_Float16*)carve((size_t)BWIN * NHEADS * NWIN * HDIM * 2);
  _Float16* ctx = (_Float16*)carve((size_t)MROWS * CDIM * 2);

  init_kernel<<<512, 256, 0, stream>>>(qkv_w, proj_w, rpb, qkv_wt, proj_wt,
                                       biasC, kT);
  qkv_gemm<<<MTILES, 256, 0, stream>>>(x, qkv_wt, qkv_b, qd, kT, vd);
  attn_kernel<<<BWIN * NHEADS, 128, 0, stream>>>(qd, kT, vd, biasC, ctx);
  proj_gemm<<<MTILES, 256, 0, stream>>>(ctx, proj_wt, proj_b, out);
}
